// EnhancedSwinTransformerBlock_CrossAttn_20925080666667
// MI455X (gfx1250) — matrix operand order repaired and checked by reading, not compiled, not run on hardware
//
#include <hip/hip_runtime.h>
#include <hip/hip_bf16.h>

// ---------------------------------------------------------------------------
// EnhancedSwinTransformerBlock + CrossAttn for MI455X (gfx1250, wave32, WMMA)
// All GEMMs / attention matmuls use v_wmma_f32_16x16x32_bf16 with f32 accum.
// GEMM A-tiles staged to LDS via gfx1250 async-to-LDS when available.
// ---------------------------------------------------------------------------

typedef __attribute__((ext_vector_type(16))) __bf16 bf16x16;
typedef __attribute__((ext_vector_type(8)))  __bf16 bf16x8;
typedef __attribute__((ext_vector_type(8)))  float  f32x8;

#define DEVI __device__ __forceinline__

#if defined(__AMDGCN__) && __has_builtin(__builtin_amdgcn_global_load_async_to_lds_b128)
#define GEMM_ASYNC 1
#else
#define GEMM_ASYNC 0
#endif

#if GEMM_ASYNC
typedef int v4i __attribute__((vector_size(4 * sizeof(int))));
typedef __attribute__((address_space(1))) v4i v4i_g;
typedef __attribute__((address_space(3))) v4i v4i_l;
DEVI void async_cp16(const __bf16* g, __bf16* l) {
  __builtin_amdgcn_global_load_async_to_lds_b128((v4i_g*)g, (v4i_l*)l, 0, 0);
}
DEVI void async_wait0() {
#if __has_builtin(__builtin_amdgcn_s_wait_asynccnt)
  __builtin_amdgcn_s_wait_asynccnt(0);
#else
  asm volatile("s_wait_asynccnt 0x0" ::: "memory");
#endif
}
#endif

DEVI bf16x16 join8(bf16x8 lo, bf16x8 hi) {
  bf16x16 r;
#pragma unroll
  for (int i = 0; i < 8; ++i) { r[i] = lo[i]; r[8 + i] = hi[i]; }
  return r;
}

DEVI f32x8 wmma_bf16(bf16x16 a, bf16x16 b, f32x8 c) {
  return __builtin_amdgcn_wmma_f32_16x16x32_bf16(false, a, false, b,
                                                 (short)0, c, false, false);
}

// 16-lane-group reductions (rows live in independent lane halves per WMMA C layout)
DEVI float red_max16(float v) {
#pragma unroll
  for (int m = 1; m <= 8; m <<= 1) v = fmaxf(v, __shfl_xor(v, m, 32));
  return v;
}
DEVI float red_sum16(float v) {
#pragma unroll
  for (int m = 1; m <= 8; m <<= 1) v += __shfl_xor(v, m, 32);
  return v;
}
DEVI float red_sum32(float v) {
#pragma unroll
  for (int m = 1; m <= 16; m <<= 1) v += __shfl_xor(v, m, 32);
  return v;
}

// shifted-window row map: window-ordered row index -> spatial token index
// B=2, S=16, H=W=32, WS=4, SHIFT=2, NW=256 per batch.
DEVI int win_map(int m) {
  int w  = m >> 6, n = m & 63;
  int b  = w >> 8; int wi = w & 255;
  int sb = wi >> 6, hb = (wi >> 3) & 7, wb = wi & 7;
  int ls = n >> 4,  lh = (n >> 2) & 3, lw = n & 3;
  int s  = (sb * 4 + ls + 2) & 15;
  int hh = (hb * 4 + lh + 2) & 31;
  int ww = (wb * 4 + lw + 2) & 31;
  return b * 16384 + s * 1024 + hh * 32 + ww;
}

DEVI int cls16(int c) { return c < 12 ? 0 : (c < 14 ? 1 : 2); }   // S dim
DEVI int cls32(int c) { return c < 28 ? 0 : (c < 30 ? 1 : 2); }   // H/W dims

// ---------------------------------------------------------------------------
// LayerNorm over C=384, fp32 in -> bf16 out. One wave per token (12 elem/lane).
// ---------------------------------------------------------------------------
__global__ __launch_bounds__(256)
void ln_kernel(const float* __restrict__ x, const float* __restrict__ w,
               const float* __restrict__ b, __bf16* __restrict__ out) {
  const int token = blockIdx.x * 8 + (threadIdx.x >> 5);
  const int lane  = threadIdx.x & 31;
  const float* p  = x + (size_t)token * 384;
  float v[12], s = 0.f, sq = 0.f;
#pragma unroll
  for (int j = 0; j < 12; ++j) {
    v[j] = p[lane + 32 * j];
    s += v[j]; sq += v[j] * v[j];
  }
  s  = red_sum32(s);
  sq = red_sum32(sq);
  const float mu  = s * (1.f / 384.f);
  const float var = sq * (1.f / 384.f) - mu * mu;
  const float rs  = rsqrtf(var + 1e-5f);
  __bf16* o = out + (size_t)token * 384;
#pragma unroll
  for (int j = 0; j < 12; ++j) {
    int c = lane + 32 * j;
    o[c] = (__bf16)((v[j] - mu) * rs * w[c] + b[c]);
  }
}

// fp32 -> bf16 conversion (for cross_features)
__global__ __launch_bounds__(256)
void cvt_kernel(const float* __restrict__ in, __bf16* __restrict__ out, int n) {
  int i = blockIdx.x * 256 + threadIdx.x;
  if (i < n) out[i] = (__bf16)in[i];
}

// Precompute relative-position bias table btab[head][n][m] (12*64*64 floats).
__global__ __launch_bounds__(256)
void bias_kernel(const float* __restrict__ rpb, float* __restrict__ btab) {
  int i = blockIdx.x * 256 + threadIdx.x;   // 49152 total
  int head = i >> 12;
  int nm = i & 4095;
  int n = nm >> 6, m = nm & 63;
  int s1 = n >> 4, h1 = (n >> 2) & 3, w1 = n & 3;
  int s2 = m >> 4, h2 = (m >> 2) & 3, w2 = m & 3;
  int idx = (s1 - s2 + 3) * 49 + (h1 - h2 + 3) * 7 + (w1 - w2 + 3);
  btab[i] = rpb[idx * 12 + head];
}

// ---------------------------------------------------------------------------
// Generic WMMA GEMM: out[M,N] = act(A[M,K](bf16) @ W[N,K]^T(fp32->bf16) + bias)
// Block tile 128x64, 8 waves in 4(M)x2(N), each wave 32x32 => 4 WMMA / K-step.
// EPI: 0 = store bf16 ; 1 = GELU(exact) + store bf16 ; 2 = fp32 out = res + acc
// AMAP: window-gather A rows ; OMAP: window-scatter out rows.
// M divisible by 128, N by 64, K by 32 (true for all call sites here).
// ---------------------------------------------------------------------------
template <int EPI, int AMAP, int OMAP>
__global__ __launch_bounds__(256)
void gemm_kernel(const __bf16* __restrict__ A, const float* __restrict__ W,
                 const float* __restrict__ bias, int N, int K,
                 __bf16* outb, const float* resf, float* outf) {
  __shared__ __bf16 As[128 * 40];
  __shared__ __bf16 Bs[64 * 40];
  const int bm = blockIdx.y * 128;
  const int bn = blockIdx.x * 64;
  const int t = threadIdx.x;
  const int wave = t >> 5, lane = t & 31;
  const int wm = (wave >> 1) * 32;       // 0,32,64,96
  const int wn = (wave & 1) * 32;        // 0,32

  f32x8 acc00 = {}, acc01 = {}, acc10 = {}, acc11 = {};

  // cooperative A load: thread t covers row t>>1 (0..127), 16 cols at (t&1)*16
  const int ar = t >> 1, ac = (t & 1) * 16;
  int arow = bm + ar;
  if (AMAP) arow = win_map(arow);
  const __bf16* aptr = A + (size_t)arow * K + ac;
  __bf16* aldst = &As[ar * 40 + ac];
  // cooperative B load: thread t covers row t>>2 (0..63), 8 cols at (t&3)*8
  const int br = t >> 2, bc = (t & 3) * 8;
  const float* wptr = W + (size_t)(bn + br) * K + bc;

  for (int kk = 0; kk < K; kk += 32) {
    float4 w0 = *(const float4*)(wptr + kk);
    float4 w1 = *(const float4*)(wptr + kk + 4);
    if (kk + 32 < K) {
      __builtin_prefetch(wptr + kk + 32, 0, 1);
      __builtin_prefetch(aptr + kk + 32, 0, 1);
    }
    bf16x8 bw;
    bw[0] = (__bf16)w0.x; bw[1] = (__bf16)w0.y; bw[2] = (__bf16)w0.z; bw[3] = (__bf16)w0.w;
    bw[4] = (__bf16)w1.x; bw[5] = (__bf16)w1.y; bw[6] = (__bf16)w1.z; bw[7] = (__bf16)w1.w;
#if GEMM_ASYNC
    __syncthreads();                       // previous tile fully consumed
    async_cp16(aptr + kk, aldst);          // A tile -> LDS via ASYNCcnt
    async_cp16(aptr + kk + 8, aldst + 8);
    *(bf16x8*)&Bs[br * 40 + bc] = bw;
    async_wait0();
    __syncthreads();
#else
    bf16x8 av0 = *(const bf16x8*)(aptr + kk);
    bf16x8 av1 = *(const bf16x8*)(aptr + kk + 8);
    __syncthreads();
    *(bf16x8*)aldst = av0;
    *(bf16x8*)(aldst + 8) = av1;
    *(bf16x8*)&Bs[br * 40 + bc] = bw;
    __syncthreads();
#endif

    const __bf16* a0p = &As[(wm + (lane & 15)) * 40 + ((lane >> 4) << 4)];
    const __bf16* a1p = a0p + 16 * 40;
    bf16x16 af0 = join8(*(const bf16x8*)a0p, *(const bf16x8*)(a0p + 8));
    bf16x16 af1 = join8(*(const bf16x8*)a1p, *(const bf16x8*)(a1p + 8));
    const __bf16* bb0 = &Bs[(wn + (lane & 15)) * 40 + ((lane >> 4) << 4)];
    const __bf16* bb1 = bb0 + 16 * 40;
    bf16x16 bf0 = join8(*(const bf16x8*)bb0, *(const bf16x8*)(bb0 + 8));
    bf16x16 bf1 = join8(*(const bf16x8*)bb1, *(const bf16x8*)(bb1 + 8));
    acc00 = wmma_bf16(af0, bf0, acc00);
    acc01 = wmma_bf16(af0, bf1, acc01);
    acc10 = wmma_bf16(af1, bf0, acc10);
    acc11 = wmma_bf16(af1, bf1, acc11);
  }

  // epilogue: C layout => row = r + 8*(lane>=16), col = lane&15 per 16x16 tile
  const int colBase = bn + wn + (lane & 15);
  const float b0 = bias[colBase];
  const float b1 = bias[colBase + 16];
#pragma unroll
  for (int i = 0; i < 2; ++i) {
    const int rbase = bm + wm + i * 16 + ((lane >> 4) << 3);
    f32x8 a0 = i ? acc10 : acc00;
    f32x8 a1 = i ? acc11 : acc01;
#pragma unroll
    for (int r = 0; r < 8; ++r) {
      int row  = rbase + r;
      int orow = OMAP ? win_map(row) : row;
      float v0 = a0[r] + b0;
      float v1 = a1[r] + b1;
      if (EPI == 1) {
        v0 = 0.5f * v0 * (1.f + erff(v0 * 0.70710678118654752f));
        v1 = 0.5f * v1 * (1.f + erff(v1 * 0.70710678118654752f));
      }
      if (EPI <= 1) {
        outb[(size_t)orow * N + colBase]      = (__bf16)v0;
        outb[(size_t)orow * N + colBase + 16] = (__bf16)v1;
      } else {
        size_t o0 = (size_t)orow * N + colBase;
        size_t o1 = o0 + 16;
        outf[o0] = resf[o0] + v0;
        outf[o1] = resf[o1] + v1;
      }
    }
  }
}

// ---------------------------------------------------------------------------
// Shifted-window attention. One block per (window w in [0,512), head in [0,12)).
// qkv layout: [(w*64+tok), 1152] bf16, col = t*384 + head*32 + hd.
// S = (Q*scale)@K^T + rel_pos_bias + shift_mask ; softmax ; O = P@V.
// 4 waves; wave owns a 16-row strip (4 S tiles of 16x16, K=32 => 1 WMMA each).
// ---------------------------------------------------------------------------
__global__ __launch_bounds__(128)
void winattn_kernel(const __bf16* __restrict__ qkv, const float* __restrict__ btab,
                    __bf16* __restrict__ out) {
  __shared__ __bf16 Vt[32 * 72];          // [hd][tok], stride 72
  __shared__ __bf16 Pst[4][16 * 72];      // per-wave P strip [16][64], stride 72

  const int blk = blockIdx.x;
  const int w = blk / 12, head = blk % 12;
  const int t = threadIdx.x, wave = t >> 5, lane = t & 31;
  const int wi = w & 255;
  const int sb = wi >> 6, hb = (wi >> 3) & 7, wb = wi & 7;
  const size_t base = (size_t)w * 64 * 1152 + head * 32;

  // stage V transposed into LDS: 128 threads, tok = t>>1, 16 hd each
  {
    int tok = t >> 1; int h0 = (t & 1) * 16;
    const __bf16* vp = qkv + base + (size_t)tok * 1152 + 768 + h0;
    bf16x8 a = *(const bf16x8*)vp;
    bf16x8 c = *(const bf16x8*)(vp + 8);
#pragma unroll
    for (int i = 0; i < 8; ++i) {
      Vt[(h0 + i) * 72 + tok]     = a[i];
      Vt[(h0 + 8 + i) * 72 + tok] = c[i];
    }
  }
  __syncthreads();

  const int n0 = wave * 16;
  const int grp = (lane >> 4) << 3;       // 0 or 8: row offset inside 16x16 tile
  const int mcol = lane & 15;
  const float scale = 0.17677669529663687f;

  // Q A-fragment (16x32, K = head dim 32)
  const __bf16* qp = qkv + base + (size_t)(n0 + mcol) * 1152 + ((lane >> 4) << 4);
  bf16x16 qa = join8(*(const bf16x8*)qp, *(const bf16x8*)(qp + 8));

  f32x8 sacc[4];
#pragma unroll
  for (int ct = 0; ct < 4; ++ct) {
    const __bf16* kp = qkv + base + (size_t)(ct * 16 + mcol) * 1152 + 384
                     + ((lane >> 4) << 4);
    bf16x16 kb = join8(*(const bf16x8*)kp, *(const bf16x8*)(kp + 8));
    f32x8 z = {};
    sacc[ct] = wmma_bf16(qa, kb, z);
  }

  // bias + mask + softmax (rows live in 16-lane halves)
  int clsm[4];
#pragma unroll
  for (int ct = 0; ct < 4; ++ct) {
    int m = ct * 16 + mcol;
    clsm[ct] = cls16(sb * 4 + (m >> 4)) * 9 + cls32(hb * 4 + ((m >> 2) & 3)) * 3
             + cls32(wb * 4 + (m & 3));
  }
  const float* brow = btab + head * 4096 + mcol;
  float rsum[8];
  __bf16* Pw = &Pst[wave][0];
#pragma unroll
  for (int r = 0; r < 8; ++r) {
    int n = n0 + r + grp;
    int clsn = cls16(sb * 4 + (n >> 4)) * 9 + cls32(hb * 4 + ((n >> 2) & 3)) * 3
             + cls32(wb * 4 + (n & 3));
    const float* bp = brow + n * 64;
    float sv[4];
    float mx = -1e30f;
#pragma unroll
    for (int ct = 0; ct < 4; ++ct) {
      float v = sacc[ct][r] * scale + bp[ct * 16];
      if (clsn != clsm[ct]) v -= 100.f;
      sv[ct] = v;
      mx = fmaxf(mx, v);
    }
    mx = red_max16(mx);
    float sm = 0.f;
#pragma unroll
    for (int ct = 0; ct < 4; ++ct) {
      float e = __expf(sv[ct] - mx);
      sm += e;
      Pw[(r + grp) * 72 + ct * 16 + mcol] = (__bf16)e;
    }
    rsum[r] = red_sum16(sm);
  }

  // O = P @ V  (P 16x64 from LDS strip, V^T from LDS)
  f32x8 o0 = {}, o1 = {};
#pragma unroll
  for (int kt = 0; kt < 2; ++kt) {
    const __bf16* pp = &Pw[mcol * 72 + kt * 32 + ((lane >> 4) << 4)];
    bf16x16 pa = join8(*(const bf16x8*)pp, *(const bf16x8*)(pp + 8));
    const __bf16* v0 = &Vt[(0 + mcol) * 72 + kt * 32 + ((lane >> 4) << 4)];
    const __bf16* v1 = &Vt[(16 + mcol) * 72 + kt * 32 + ((lane >> 4) << 4)];
    bf16x16 vb0 = join8(*(const bf16x8*)v0, *(const bf16x8*)(v0 + 8));
    bf16x16 vb1 = join8(*(const bf16x8*)v1, *(const bf16x8*)(v1 + 8));
    o0 = wmma_bf16(pa, vb0, o0);
    o1 = wmma_bf16(pa, vb1, o1);
  }

#pragma unroll
  for (int r = 0; r < 8; ++r) {
    int tokrow = w * 64 + n0 + r + grp;
    float inv = 1.f / rsum[r];
    out[(size_t)tokrow * 384 + head * 32 + mcol]      = (__bf16)(o0[r] * inv);
    out[(size_t)tokrow * 384 + head * 32 + 16 + mcol] = (__bf16)(o1[r] * inv);
  }
}

// ---------------------------------------------------------------------------
// Cross attention. One block per (b, head, 64-query tile). 512 keys, online
// softmax over 8 chunks of 64. K and V^T fully staged in LDS.
// qc: [b*16384+tok, 384] bf16 ; kvc: [b*512+tok, 768] bf16 (k at +0, v at +384)
// ---------------------------------------------------------------------------
__global__ __launch_bounds__(128)
void crossattn_kernel(const __bf16* __restrict__ qc, const __bf16* __restrict__ kvc,
                      __bf16* __restrict__ out) {
  __shared__ __bf16 Ks[512 * 40];         // [tok][hd], stride 40
  __shared__ __bf16 Vt[32 * 520];         // [hd][tok], stride 520
  __shared__ __bf16 Pst[4][16 * 72];

  const int blk = blockIdx.x;
  const int qt = blk & 255;
  const int head = (blk >> 8) % 12;
  const int b = blk / (256 * 12);
  const int t = threadIdx.x, wave = t >> 5, lane = t & 31;
  const size_t kvbase = (size_t)b * 512 * 768 + head * 32;

  // stage K (natural) and V (transposed)
#pragma unroll
  for (int i = 0; i < 4; ++i) {
    int tok = t * 4 + i;
    const __bf16* kp = kvc + kvbase + (size_t)tok * 768;
    bf16x8 k0 = *(const bf16x8*)kp;
    bf16x8 k1 = *(const bf16x8*)(kp + 8);
    bf16x8 k2 = *(const bf16x8*)(kp + 16);
    bf16x8 k3 = *(const bf16x8*)(kp + 24);
    *(bf16x8*)&Ks[tok * 40 + 0]  = k0;
    *(bf16x8*)&Ks[tok * 40 + 8]  = k1;
    *(bf16x8*)&Ks[tok * 40 + 16] = k2;
    *(bf16x8*)&Ks[tok * 40 + 24] = k3;
    const __bf16* vp = kvc + kvbase + (size_t)tok * 768 + 384;
    bf16x8 vv[4];
    vv[0] = *(const bf16x8*)vp;
    vv[1] = *(const bf16x8*)(vp + 8);
    vv[2] = *(const bf16x8*)(vp + 16);
    vv[3] = *(const bf16x8*)(vp + 24);
#pragma unroll
    for (int c = 0; c < 4; ++c)
#pragma unroll
      for (int j = 0; j < 8; ++j)
        Vt[(c * 8 + j) * 520 + tok] = vv[c][j];
  }
  __syncthreads();

  const int m0 = qt * 64 + wave * 16;
  const int grp = (lane >> 4) << 3;
  const int mcol = lane & 15;
  const float scale = 0.17677669529663687f;

  const __bf16* qp = qc + (size_t)(b * 16384 + m0 + mcol) * 384 + head * 32 + ((lane >> 4) << 4);
  bf16x16 qa = join8(*(const bf16x8*)qp, *(const bf16x8*)(qp + 8));

  float mrow[8], rsum[8];
#pragma unroll
  for (int r = 0; r < 8; ++r) { mrow[r] = -1e30f; rsum[r] = 0.f; }
  f32x8 o0 = {}, o1 = {};
  __bf16* Pw = &Pst[wave][0];

  for (int kc = 0; kc < 8; ++kc) {
    f32x8 sacc[4];
#pragma unroll
    for (int ct = 0; ct < 4; ++ct) {
      const __bf16* kp = &Ks[(kc * 64 + ct * 16 + mcol) * 40 + ((lane >> 4) << 4)];
      bf16x16 kb = join8(*(const bf16x8*)kp, *(const bf16x8*)(kp + 8));
      f32x8 z = {};
      sacc[ct] = wmma_bf16(qa, kb, z);
    }
#pragma unroll
    for (int r = 0; r < 8; ++r) {
      float mx = -1e30f;
#pragma unroll
      for (int ct = 0; ct < 4; ++ct) mx = fmaxf(mx, sacc[ct][r] * scale);
      mx = red_max16(mx);
      float mnew = fmaxf(mrow[r], mx);
      float corr = __expf(mrow[r] - mnew);
      mrow[r] = mnew;
      float sm = 0.f;
#pragma unroll
      for (int ct = 0; ct < 4; ++ct) {
        float e = __expf(sacc[ct][r] * scale - mnew);
        sm += e;
        Pw[(r + grp) * 72 + ct * 16 + mcol] = (__bf16)e;
      }
      rsum[r] = rsum[r] * corr + red_sum16(sm);
      o0[r] = o0[r] * corr;
      o1[r] = o1[r] * corr;
    }
#pragma unroll
    for (int kt = 0; kt < 2; ++kt) {
      const __bf16* pp = &Pw[mcol * 72 + kt * 32 + ((lane >> 4) << 4)];
      bf16x16 pa = join8(*(const bf16x8*)pp, *(const bf16x8*)(pp + 8));
      const __bf16* v0 = &Vt[(0 + mcol) * 520 + kc * 64 + kt * 32 + ((lane >> 4) << 4)];
      const __bf16* v1 = &Vt[(16 + mcol) * 520 + kc * 64 + kt * 32 + ((lane >> 4) << 4)];
      bf16x16 vb0 = join8(*(const bf16x8*)v0, *(const bf16x8*)(v0 + 8));
      bf16x16 vb1 = join8(*(const bf16x8*)v1, *(const bf16x8*)(v1 + 8));
      o0 = wmma_bf16(pa, vb0, o0);
      o1 = wmma_bf16(pa, vb1, o1);
    }
  }

#pragma unroll
  for (int r = 0; r < 8; ++r) {
    int row = b * 16384 + m0 + r + grp;
    float inv = 1.f / rsum[r];
    out[(size_t)row * 384 + head * 32 + mcol]      = (__bf16)(o0[r] * inv);
    out[(size_t)row * 384 + head * 32 + 16 + mcol] = (__bf16)(o1[r] * inv);
  }
}

// ---------------------------------------------------------------------------
// Orchestration
// ---------------------------------------------------------------------------
extern "C" void kernel_launch(void* const* d_in, const int* in_sizes, int n_in,
                              void* d_out, int out_size, void* d_ws, size_t ws_size,
                              hipStream_t stream) {
  (void)in_sizes; (void)n_in; (void)out_size; (void)ws_size;
  const float* x_in    = (const float*)d_in[0];
  const float* crossf  = (const float*)d_in[1];
  const float* n1w     = (const float*)d_in[2];
  const float* n1b     = (const float*)d_in[3];
  const float* qkv_w   = (const float*)d_in[4];
  const float* qkv_b   = (const float*)d_in[5];
  const float* rpb     = (const float*)d_in[6];
  const float* proj_w  = (const float*)d_in[7];
  const float* proj_b  = (const float*)d_in[8];
  const float* ncw     = (const float*)d_in[9];
  const float* ncb     = (const float*)d_in[10];
  const float* q_w     = (const float*)d_in[11];
  const float* q_b     = (const float*)d_in[12];
  const float* kv_w    = (const float*)d_in[13];
  const float* kv_b    = (const float*)d_in[14];
  const float* cproj_w = (const float*)d_in[15];
  const float* cproj_b = (const float*)d_in[16];
  const float* n2w     = (const float*)d_in[17];
  const float* n2b     = (const float*)d_in[18];
  const float* fc1_w   = (const float*)d_in[19];
  const float* fc1_b   = (const float*)d_in[20];
  const float* fc2_w   = (const float*)d_in[21];
  const float* fc2_b   = (const float*)d_in[22];

  char* ws = (char*)d_ws;
  float*  xbuf = (float*)ws;   ws += 32768ull * 384 * 4;   // running residual
  __bf16* xln  = (__bf16*)ws;  ws += 32768ull * 384 * 2;   // LN outputs (reused)
  __bf16* big  = (__bf16*)ws;  ws += 32768ull * 1536 * 2;  // qkv / cf_bf16 / mlp hidden
  __bf16* bufB = (__bf16*)ws;  ws += 32768ull * 384 * 2;   // attn_out / cross out
  __bf16* bufC = (__bf16*)ws;  ws += 32768ull * 384 * 2;   // q (cross)
  __bf16* kvcb = (__bf16*)ws;  ws += 1024ull * 768 * 2;    // kv projection
  float*  btab = (float*)ws;   ws += 12ull * 64 * 64 * 4;  // rel-pos bias table

  // --- stage 1: shifted-window attention ---
  bias_kernel<<<192, 256, 0, stream>>>(rpb, btab);
  ln_kernel<<<4096, 256, 0, stream>>>(x_in, n1w, n1b, xln);
  gemm_kernel<0, 1, 0><<<dim3(1152 / 64, 32768 / 128), 256, 0, stream>>>(
      xln, qkv_w, qkv_b, 1152, 384, big, nullptr, nullptr);
  winattn_kernel<<<512 * 12, 128, 0, stream>>>(big, btab, bufB);
  gemm_kernel<2, 0, 1><<<dim3(384 / 64, 32768 / 128), 256, 0, stream>>>(
      bufB, proj_w, proj_b, 384, 384, nullptr, x_in, xbuf);

  // --- stage 2: cross attention ---
  ln_kernel<<<4096, 256, 0, stream>>>(xbuf, ncw, ncb, xln);
  gemm_kernel<0, 0, 0><<<dim3(384 / 64, 32768 / 128), 256, 0, stream>>>(
      xln, q_w, q_b, 384, 384, bufC, nullptr, nullptr);
  cvt_kernel<<<(196608 + 255) / 256, 256, 0, stream>>>(crossf, big, 196608);
  gemm_kernel<0, 0, 0><<<dim3(768 / 64, 1024 / 128), 256, 0, stream>>>(
      big, kv_w, kv_b, 768, 192, kvcb, nullptr, nullptr);
  crossattn_kernel<<<2 * 12 * 256, 128, 0, stream>>>(bufC, kvcb, bufB);
  gemm_kernel<2, 0, 0><<<dim3(384 / 64, 32768 / 128), 256, 0, stream>>>(
      bufB, cproj_w, cproj_b, 384, 384, nullptr, xbuf, xbuf);

  // --- stage 3: MLP ---
  ln_kernel<<<4096, 256, 0, stream>>>(xbuf, n2w, n2b, xln);
  gemm_kernel<1, 0, 0><<<dim3(1536 / 64, 32768 / 128), 256, 0, stream>>>(
      xln, fc1_w, fc1_b, 1536, 384, big, nullptr, nullptr);
  gemm_kernel<2, 0, 0><<<dim3(384 / 64, 32768 / 128), 256, 0, stream>>>(
      big, fc2_w, fc2_b, 384, 1536, nullptr, xbuf, (float*)d_out);
}
